// SelfAttention_11227044511836
// MI455X (gfx1250) — matrix operand order repaired and checked by reading, not compiled, not run on hardware
//
#include <hip/hip_runtime.h>

#define EMBED 1024
#define SEQ   2048
#define BATCH 4
#define MTOT  (BATCH * SEQ)   // 8192

typedef __attribute__((ext_vector_type(8)))  float          f32x8;
typedef __attribute__((ext_vector_type(16))) __bf16         bf16x16;
typedef __attribute__((ext_vector_type(8)))  unsigned short u16x8;
typedef __attribute__((ext_vector_type(16))) unsigned short u16x16;
typedef __attribute__((ext_vector_type(4)))  unsigned int   u32x4;
typedef __attribute__((ext_vector_type(4)))  int            i32x4;
typedef __attribute__((ext_vector_type(8)))  int            i32x8;

union Frag {
    u16x16  u;
    u16x8   h[2];
    bf16x16 bf;
};

// float -> bf16 bits, round-to-nearest-even
__device__ __forceinline__ unsigned short f2bf(float f) {
    unsigned int u = __float_as_uint(f);
    u += 0x7FFFu + ((u >> 16) & 1u);
    return (unsigned short)(u >> 16);
}

// ---------------------------------------------------------------------------
// TDM: issue a 2D tile load (rows x tile_k bf16 elements, row stride
// row_stride elements) from global memory into LDS at byte offset lds_off.
// D# built per CDNA5 ISA (group0/group1).
// Tracked by TENSORcnt. data_size=1 (2 bytes). tile_dim0*2B is a multiple
// of 4 bytes as required. Tensor dims set large so OOB clamping never fires.
// ---------------------------------------------------------------------------
__device__ __forceinline__ void tdm_load_2d_to_lds(
    unsigned lds_off, const void* gptr,
    unsigned tile_k, unsigned tile_rows, unsigned row_stride)
{
    unsigned long long ga = (unsigned long long)(uintptr_t)gptr;
    const unsigned td0 = 1u << 20;   // tensor_dim0 (OOB bound, never clamps)
    const unsigned td1 = 1u << 20;   // tensor_dim1

    u32x4 g0;
    g0[0] = 1u;                                   // count=1, user mode
    g0[1] = lds_off;                              // lds_addr [63:32]
    g0[2] = (unsigned)(ga & 0xFFFFFFFFu);         // global_addr lo
    g0[3] = ((unsigned)((ga >> 32) & 0x01FFFFFFu))
          | 0x80000000u;                          // global_addr hi | type=2

    i32x8 g1;
    g1[0] = (int)(1u << 16);                      // data_size=1 (2B); mask=0
    g1[1] = (int)((td0 & 0xFFFFu) << 16);         // tensor_dim0[15:0]
    g1[2] = (int)(((td0 >> 16) & 0xFFFFu) | ((td1 & 0xFFFFu) << 16));
    g1[3] = (int)(((td1 >> 16) & 0xFFFFu) | ((tile_k & 0xFFFFu) << 16)); // tile_dim0
    g1[4] = (int)(tile_rows & 0xFFFFu);           // tile_dim1; tile_dim2=0
    g1[5] = (int)row_stride;                      // tensor_dim0_stride[31:0]
    g1[6] = 0;                                    // stride hi / dim1_stride
    g1[7] = 0;

    i32x4 z4 = {0, 0, 0, 0};
#if defined(__clang_major__) && (__clang_major__ >= 23)
    i32x8 z8 = {0, 0, 0, 0, 0, 0, 0, 0};
    __builtin_amdgcn_tensor_load_to_lds(g0, g1, z4, z4, z8, 0);
#else
    __builtin_amdgcn_tensor_load_to_lds(g0, g1, z4, z4, 0);
#endif
}

// ---------------------------------------------------------------------------
// fp32 -> bf16 conversion (grid-stride)
// ---------------------------------------------------------------------------
__global__ void __launch_bounds__(256) convert_bf16_kernel(
    const float* __restrict__ in, unsigned short* __restrict__ out, int n)
{
    int i = blockIdx.x * blockDim.x + threadIdx.x;
    int stride = gridDim.x * blockDim.x;
    for (; i < n; i += stride) out[i] = f2bf(in[i]);
}

// ---------------------------------------------------------------------------
// Generic bf16 WMMA GEMM:  C[m][n] = sum_k A[m][k] * B[n][k]
//   A: rows over M, contiguous K (ldA);  B: rows over N, contiguous K (ldB)
// Block tile 256(M) x 128(N); 8 waves in 4(M) x 2(N); 64x64 per wave.
// K staged 32 at a time through double-buffered LDS, filled by wave 0 via
// TDM tensor_load_to_lds, gated by s_wait_tensorcnt + one barrier per step.
// mode 0: bf16 row-major   mode 1: bf16 transposed C[n*ldC+m]   mode 2: f32*scale
// ---------------------------------------------------------------------------
__global__ void __launch_bounds__(256) wmma_gemm_kernel(
    const unsigned short* __restrict__ A, size_t strideAz, int ldA,
    const unsigned short* __restrict__ B, size_t strideBz, int ldB,
    void* __restrict__ C, size_t strideCz, int ldC,
    int K, int mode, float scale)
{
    __shared__ unsigned short sA[2][256 * 32];   // 16 KB per buffer
    __shared__ unsigned short sB[2][128 * 32];   //  8 KB per buffer

    const int lane = threadIdx.x & 31;
    const int wave = threadIdx.x >> 5;
    const int wm   = wave >> 1;      // 0..3 : wave row
    const int wn   = wave & 1;       // 0..1 : wave col
    const int lrow = lane & 15;      // A row / B,C col within 16-tile
    const int lhi  = lane >> 4;      // lane half

    const int m0 = blockIdx.x * 256;
    const int n0 = blockIdx.y * 128;
    const int z  = blockIdx.z;

    const unsigned short* Az = A + (size_t)z * strideAz;
    const unsigned short* Bz = B + (size_t)z * strideBz;

    f32x8 acc[4][4] = {};

    // preload k0 = 0 into buffer 0
    if (wave == 0) {
        tdm_load_2d_to_lds((unsigned)(uintptr_t)&sA[0][0],
                           Az + (size_t)m0 * ldA, 32, 256, (unsigned)ldA);
        tdm_load_2d_to_lds((unsigned)(uintptr_t)&sB[0][0],
                           Bz + (size_t)n0 * ldB, 32, 128, (unsigned)ldB);
    }

    int buf = 0;
    for (int k0 = 0; k0 < K; k0 += 32, buf ^= 1) {
        if (wave == 0) __builtin_amdgcn_s_wait_tensorcnt(0);
        __syncthreads();   // buf ready for all waves; prior reads of buf^1 done
        if (wave == 0 && (k0 + 32) < K) {
            tdm_load_2d_to_lds((unsigned)(uintptr_t)&sA[buf ^ 1][0],
                               Az + (size_t)m0 * ldA + (k0 + 32), 32, 256,
                               (unsigned)ldA);
            tdm_load_2d_to_lds((unsigned)(uintptr_t)&sB[buf ^ 1][0],
                               Bz + (size_t)n0 * ldB + (k0 + 32), 32, 128,
                               (unsigned)ldB);
        }

        Frag a[4], b[4];
        #pragma unroll
        for (int t = 0; t < 4; ++t) {
            const unsigned short* pa =
                &sA[buf][(wm * 64 + t * 16 + lrow) * 32 + lhi * 8];
            a[t].h[0] = *(const u16x8*)(pa);
            a[t].h[1] = *(const u16x8*)(pa + 16);
        }
        #pragma unroll
        for (int u = 0; u < 4; ++u) {
            const unsigned short* pb =
                &sB[buf][(wn * 64 + u * 16 + lrow) * 32 + lhi * 8];
            b[u].h[0] = *(const u16x8*)(pb);
            b[u].h[1] = *(const u16x8*)(pb + 16);
        }
        #pragma unroll
        for (int t = 0; t < 4; ++t)
            #pragma unroll
            for (int u = 0; u < 4; ++u)
                acc[t][u] = __builtin_amdgcn_wmma_f32_16x16x32_bf16(
                    false, a[t].bf, false, b[u].bf,
                    (short)0, acc[t][u], false, false);
    }

    // D layout: element r of each lane is (M = r + 8*lhi, N = lrow) in-tile
    if (mode == 0) {
        unsigned short* Cu = (unsigned short*)C + (size_t)z * strideCz;
        #pragma unroll
        for (int t = 0; t < 4; ++t)
            #pragma unroll
            for (int u = 0; u < 4; ++u) {
                const int m = m0 + wm * 64 + t * 16 + lhi * 8;
                const int n = n0 + wn * 64 + u * 16 + lrow;
                #pragma unroll
                for (int r = 0; r < 8; ++r)
                    Cu[(size_t)(m + r) * ldC + n] = f2bf(acc[t][u][r]);
            }
    } else if (mode == 1) {
        unsigned short* Cu = (unsigned short*)C + (size_t)z * strideCz;
        #pragma unroll
        for (int t = 0; t < 4; ++t)
            #pragma unroll
            for (int u = 0; u < 4; ++u) {
                const int n = n0 + wn * 64 + u * 16 + lrow;   // transposed row
                const int m = m0 + wm * 64 + t * 16 + lhi * 8;
                u16x8 v;
                #pragma unroll
                for (int r = 0; r < 8; ++r) v[r] = f2bf(acc[t][u][r]);
                *(u16x8*)(Cu + (size_t)n * ldC + m) = v;
            }
    } else {
        float* Cf = (float*)C + (size_t)z * strideCz;
        #pragma unroll
        for (int t = 0; t < 4; ++t)
            #pragma unroll
            for (int u = 0; u < 4; ++u) {
                const int m = m0 + wm * 64 + t * 16 + lhi * 8;
                const int n = n0 + wn * 64 + u * 16 + lrow;
                #pragma unroll
                for (int r = 0; r < 8; ++r)
                    Cf[(size_t)(m + r) * ldC + n] = acc[t][u][r] * scale;
            }
    }
}

// ---------------------------------------------------------------------------
// Row softmax: one 256-thread block per row of 2048 fp32, output bf16
// ---------------------------------------------------------------------------
__global__ void __launch_bounds__(256) softmax_kernel(
    const float* __restrict__ En, unsigned short* __restrict__ P)
{
    const int row = blockIdx.x;
    const int tid = threadIdx.x;
    const float* e = En + (size_t)row * SEQ;

    float v[8];
    float mx = -3.402823466e38f;
    #pragma unroll
    for (int i = 0; i < 8; ++i) {
        v[i] = e[tid + i * 256];
        mx = fmaxf(mx, v[i]);
    }

    __shared__ float red[256];
    red[tid] = mx;
    __syncthreads();
    #pragma unroll
    for (int s = 128; s > 0; s >>= 1) {
        if (tid < s) red[tid] = fmaxf(red[tid], red[tid + s]);
        __syncthreads();
    }
    mx = red[0];
    __syncthreads();

    float sum = 0.f;
    #pragma unroll
    for (int i = 0; i < 8; ++i) {
        v[i] = __expf(v[i] - mx);
        sum += v[i];
    }
    red[tid] = sum;
    __syncthreads();
    #pragma unroll
    for (int s = 128; s > 0; s >>= 1) {
        if (tid < s) red[tid] += red[tid + s];
        __syncthreads();
    }
    const float inv = 1.0f / red[0];

    unsigned short* p = P + (size_t)row * SEQ;
    #pragma unroll
    for (int i = 0; i < 8; ++i) p[tid + i * 256] = f2bf(v[i] * inv);
}

// ---------------------------------------------------------------------------
extern "C" void kernel_launch(void* const* d_in, const int* in_sizes, int n_in,
                              void* d_out, int out_size, void* d_ws, size_t ws_size,
                              hipStream_t stream) {
    const float* x  = (const float*)d_in[0];
    const float* Wq = (const float*)d_in[1];
    const float* Wk = (const float*)d_in[2];
    const float* Wv = (const float*)d_in[3];
    float* out = (float*)d_out;

    char* ws = (char*)d_ws;
    size_t off = 0;
    unsigned short* Xbf  = (unsigned short*)(ws + off); off += (size_t)MTOT * EMBED * 2;   // 16 MB
    unsigned short* Wqbf = (unsigned short*)(ws + off); off += (size_t)EMBED * EMBED * 2;  //  2 MB
    unsigned short* Wkbf = (unsigned short*)(ws + off); off += (size_t)EMBED * EMBED * 2;
    unsigned short* Wvbf = (unsigned short*)(ws + off); off += (size_t)EMBED * EMBED * 2;
    unsigned short* Qbf  = (unsigned short*)(ws + off); off += (size_t)MTOT * EMBED * 2;
    unsigned short* Kbf  = (unsigned short*)(ws + off); off += (size_t)MTOT * EMBED * 2;
    unsigned short* Vtbf = (unsigned short*)(ws + off); off += (size_t)EMBED * MTOT * 2;   // [o][m]
    float*          En   = (float*)(ws + off);          off += (size_t)BATCH * SEQ * SEQ * 4; // 64 MB
    unsigned short* P    = (unsigned short*)(ws + off); off += (size_t)BATCH * SEQ * SEQ * 2; // 32 MB

    // 1) fp32 -> bf16
    convert_bf16_kernel<<<2048, 256, 0, stream>>>(x,  Xbf,  MTOT * EMBED);
    convert_bf16_kernel<<<512,  256, 0, stream>>>(Wq, Wqbf, EMBED * EMBED);
    convert_bf16_kernel<<<512,  256, 0, stream>>>(Wk, Wkbf, EMBED * EMBED);
    convert_bf16_kernel<<<512,  256, 0, stream>>>(Wv, Wvbf, EMBED * EMBED);

    // 2) Q = X*Wq^T, K = X*Wk^T (row-major bf16), Vt = (X*Wv^T)^T  [o][m]
    dim3 g1(MTOT / 256, EMBED / 128, 1);
    wmma_gemm_kernel<<<g1, 256, 0, stream>>>(Xbf, 0, EMBED, Wqbf, 0, EMBED,
                                             Qbf, 0, EMBED, EMBED, 0, 1.0f);
    wmma_gemm_kernel<<<g1, 256, 0, stream>>>(Xbf, 0, EMBED, Wkbf, 0, EMBED,
                                             Kbf, 0, EMBED, EMBED, 0, 1.0f);
    wmma_gemm_kernel<<<g1, 256, 0, stream>>>(Xbf, 0, EMBED, Wvbf, 0, EMBED,
                                             Vtbf, 0, MTOT, EMBED, 1, 1.0f);

    // 3) energy = (Q K^T) / sqrt(1024), fp32, per batch
    dim3 g2(SEQ / 256, SEQ / 128, BATCH);
    wmma_gemm_kernel<<<g2, 256, 0, stream>>>(
        Qbf, (size_t)SEQ * EMBED, EMBED,
        Kbf, (size_t)SEQ * EMBED, EMBED,
        En,  (size_t)SEQ * SEQ,   SEQ,
        EMBED, 2, 0.03125f);

    // 4) row softmax -> P (bf16)
    softmax_kernel<<<BATCH * SEQ, 256, 0, stream>>>(En, P);

    // 5) out = P @ V  (A = P rows over k; B = Vt rows over output-col o)
    dim3 g3(SEQ / 256, EMBED / 128, BATCH);
    wmma_gemm_kernel<<<g3, 256, 0, stream>>>(
        P,    (size_t)SEQ * SEQ, SEQ,
        Vtbf, (size_t)SEQ,       MTOT,   // per-batch column offset into Vt rows
        out,  (size_t)SEQ * EMBED, EMBED,
        SEQ, 2, 1.0f);
}
